// SimpleRNN_1606317769030
// MI455X (gfx1250) — matrix operand order repaired and checked by reading, not compiled, not run on hardware
//
#include <hip/hip_runtime.h>
#include <hip/hip_bf16.h>

typedef __attribute__((ext_vector_type(16))) _Float16 v16h;
typedef __attribute__((ext_vector_type(8)))  float    v8f;

#define HID  20
#define TLEN 2048

__device__ __forceinline__ float fast_tanh(float x) {
#if __has_builtin(__builtin_amdgcn_tanhf)
  return __builtin_amdgcn_tanhf(x);
#else
  float e = __builtin_amdgcn_exp2f(x * 2.8853900817779268f);
  return 1.0f - 2.0f * __builtin_amdgcn_rcpf(e + 1.0f);
#endif
}

// Recurrence computed transposed per 16-batch tile:  h'^T = tanh(A . B~)
// with an AUGMENTED state vector B~ = [h(0..19), x_t, 1, 0...] (K=32) and
// A = [W_hh | W_ih | b_ih+b_hh] so the input projection + bias ride inside
// the WMMA (C operand is constant zero).
//
// Row permutation (D position -> hidden unit) so D lands exactly where the
// next step's B fragment needs it (no cross-lane moves):
//   tile0 rows 0..7  -> j = 0..7    (lo lanes, B k=0..7)
//   tile1 rows 0..7  -> j = 8..15   (lo lanes, B k=8..15)
//   tile0 rows 8..11 -> j = 16..19  (hi lanes, B k=16..19)
//   everything else  -> zero pad
// B augmentation slots: k=20 (x_t) and k=21 (1.0) = hi-lane elements 4,5.
__global__ __launch_bounds__(256) void rnn_wmma_kernel(
    const float* __restrict__ x,    const float* __restrict__ W_ih,
    const float* __restrict__ W_hh, const float* __restrict__ b_ih,
    const float* __restrict__ b_hh, const float* __restrict__ W_fc,
    const float* __restrict__ b_fc, float* __restrict__ out,
    int Btot, int T)
{
  // ---- stage small weights in LDS once per block ----
  __shared__ float smem[464];           // [0,400) W_hh, [400,420) W_ih,
  float* Wl   = smem;                   // [420,440) b_ih+b_hh, [440,460) W_fc, [460] b_fc
  float* wihl = smem + 400;
  float* bsl  = smem + 420;
  float* wfcl = smem + 440;
  for (int i = threadIdx.x; i < HID * HID; i += blockDim.x) Wl[i] = W_hh[i];
  if (threadIdx.x < HID) {
    wihl[threadIdx.x] = W_ih[threadIdx.x];
    bsl[threadIdx.x]  = b_ih[threadIdx.x] + b_hh[threadIdx.x];
    wfcl[threadIdx.x] = W_fc[threadIdx.x];
  }
  if (threadIdx.x == 0) smem[460] = b_fc[0];
  __syncthreads();

  const int  lane = threadIdx.x & 31;
  const int  wave = threadIdx.x >> 5;
  const bool hi   = lane >= 16;
  const int  n    = lane & 15;                       // batch column within tile
  const int  b0   = (blockIdx.x * (blockDim.x >> 5) + wave) * 16;
  if (b0 >= Btot) return;                            // wave-uniform: EXEC stays full
  const int  b    = b0 + n;

  // ---- A fragments (f16 16x32). Lane L holds A row m = L%16.
  // element e -> K = e + (hi?16:0)
  v16h a0, a1;
  const int  m  = n;                     // this lane's A row in both tiles
  const int  j0 = (m < 8) ? m : m + 8;   // tile0: rows 0..7 -> j=m; 8..11 -> j=m+8
  const bool v0 = (m < 12);
  const int  j1 = m + 8;                 // tile1: rows 0..7 -> j=m+8
  const bool v1 = (m < 8);
#pragma unroll
  for (int e = 0; e < 16; ++e) {
    int k = e + (hi ? 16 : 0);
    float w0 = 0.0f, w1 = 0.0f;
    if (k < HID) {                       // recurrent weights
      w0 = v0 ? Wl[j0 * HID + k] : 0.0f;
      w1 = v1 ? Wl[j1 * HID + k] : 0.0f;
    } else if (k == HID) {               // input projection column (x slot)
      w0 = v0 ? wihl[j0] : 0.0f;
      w1 = v1 ? wihl[j1] : 0.0f;
    } else if (k == HID + 1) {           // bias column (1.0 slot)
      w0 = v0 ? bsl[j0] : 0.0f;
      w1 = v1 ? bsl[j1] : 0.0f;
    }
    a0[e] = (_Float16)w0;
    a1[e] = (_Float16)w1;
  }

  // ---- B fragment: [h, x, 1, 0..]^T ; element i @ lane -> k = (hi?16:0)+i ----
  v16h bfrag;
#pragma unroll
  for (int i = 0; i < 16; ++i) bfrag[i] = (_Float16)0.0f;

  const v8f czero = {0.f, 0.f, 0.f, 0.f, 0.f, 0.f, 0.f, 0.f};
  const _Float16 onef = (_Float16)1.0f;

  const float* __restrict__ xrow = x + (size_t)b * (size_t)T;

  float4 xq = *(const float4*)(xrow);                 // prime double buffer
  for (int t4 = 0; t4 < T; t4 += 4) {
    float4 xn = *(const float4*)(xrow + ((t4 + 4) & (TLEN - 1)));
    __builtin_prefetch(xrow + ((t4 + 256) & (TLEN - 1)), 0, 3);
    float xs[4] = {xq.x, xq.y, xq.z, xq.w};
#pragma unroll
    for (int s = 0; s < 4; ++s) {
      _Float16 xh = (_Float16)xs[s];
      // inject input + bias-one into augmented K slots (hi lanes: k=20,21)
      bfrag[4] = hi ? xh   : bfrag[4];
      bfrag[5] = hi ? onef : bfrag[5];
      v8f d0 = __builtin_amdgcn_wmma_f32_16x16x32_f16(
          false, a0, false, bfrag, (short)0, czero, false, false);
      v8f d1 = __builtin_amdgcn_wmma_f32_16x16x32_f16(
          false, a1, false, bfrag, (short)0, czero, false, false);
      // D registers are already in B-fragment position: no cross-lane moves.
#pragma unroll
      for (int i = 0; i < 8; ++i) bfrag[i]     = (_Float16)fast_tanh(d0[i]);
#pragma unroll
      for (int i = 0; i < 8; ++i) bfrag[8 + i] = (_Float16)fast_tanh(d1[i]);
    }
    xq = xn;
  }

  // ---- final FC + sigmoid: out[b] = sigmoid(h . W_fc + b_fc) ----
  // (k >= 20, i.e. the x/1 slots, get weight 0 and drop out automatically)
  float partial = 0.0f;
  int kbase = hi ? 16 : 0;
#pragma unroll
  for (int i = 0; i < 16; ++i) {
    int k = kbase + i;
    float wf = (k < HID) ? wfcl[k] : 0.0f;
    partial = fmaf((float)bfrag[i], wf, partial);
  }
  partial += __shfl_xor(partial, 16, 32);
  if (!hi) {
    float z = partial + smem[460];
    float sig = __builtin_amdgcn_rcpf(1.0f + __builtin_amdgcn_exp2f(-z * 1.4426950408889634f));
    out[b] = sig;
  }
}

extern "C" void kernel_launch(void* const* d_in, const int* in_sizes, int n_in,
                              void* d_out, int out_size, void* d_ws, size_t ws_size,
                              hipStream_t stream) {
  const float* x    = (const float*)d_in[0];
  const float* W_ih = (const float*)d_in[1];
  const float* W_hh = (const float*)d_in[2];
  const float* b_ih = (const float*)d_in[3];
  const float* b_hh = (const float*)d_in[4];
  const float* W_fc = (const float*)d_in[5];
  const float* b_fc = (const float*)d_in[6];
  float* out = (float*)d_out;

  const int T = TLEN;
  const int B = in_sizes[0] / T;           // x is [B, T, 1]
  const int batchPerBlock = 16 * 8;        // 16 batch per wave, 8 waves per block
  const int grid = (B + batchPerBlock - 1) / batchPerBlock;

  rnn_wmma_kernel<<<grid, 256, 0, stream>>>(x, W_ih, W_hh, b_ih, b_hh,
                                            W_fc, b_fc, out, B, T);
}
